// AttentionITBlock_30709016167111
// MI455X (gfx1250) — matrix operand order repaired and checked by reading, not compiled, not run on hardware
//
#include <hip/hip_runtime.h>
#include <math.h>

// ---------------- problem constants ----------------
#define B_   8
#define C_   128
#define H_   96
#define W_   96
#define HW_  (H_*W_)          // 9216
#define M1_  24
#define M2_  24
#define S_   (M1_*M2_)        // 576
#define NH_  4
#define DH_  32

typedef __bf16 bf16x16 __attribute__((ext_vector_type(16)));
typedef float  f32x8   __attribute__((ext_vector_type(8)));

__device__ __forceinline__ f32x8 wmma_bf16(bf16x16 a, bf16x16 b, f32x8 c) {
  return __builtin_amdgcn_wmma_f32_16x16x32_bf16(false, a, false, b, (short)0, c, false, false);
}

// ---------------- fragment loaders (wave32, ISA §7.12.2 layouts) ----------------
// A 16x32 bf16: lane l: row r=l&15, half h=l>>4 ; element e -> K = (e<8 ? e : e+8) + 8h
// B 32x16 bf16: lane l: col c=l&15, half h=l>>4 ; element e -> K = (e<8 ? e : e+8) + 8h
// C/D 16x16 f32: vgpr v: row = v + 8h, col = l&15

// A where A[r][k] = base[r*ld + k]
__device__ __forceinline__ bf16x16 frag_a_row(const float* __restrict__ base, int ld) {
  const int l = threadIdx.x & 31, r = l & 15, h = l >> 4;
  const float* p = base + (size_t)r * ld + 8 * h;
  bf16x16 f;
#pragma unroll
  for (int e = 0; e < 8; ++e) { f[e] = (__bf16)p[e]; f[e + 8] = (__bf16)p[e + 16]; }
  return f;
}
// A row-major, negated
__device__ __forceinline__ bf16x16 frag_a_row_neg(const float* __restrict__ base, int ld) {
  const int l = threadIdx.x & 31, r = l & 15, h = l >> 4;
  const float* p = base + (size_t)r * ld + 8 * h;
  bf16x16 f;
#pragma unroll
  for (int e = 0; e < 8; ++e) { f[e] = (__bf16)(-p[e]); f[e + 8] = (__bf16)(-p[e + 16]); }
  return f;
}
// A row-major, per-K scale (alpha folding)
__device__ __forceinline__ bf16x16 frag_a_row_scaled(const float* __restrict__ base, int ld,
                                                     const float* __restrict__ sk) {
  const int l = threadIdx.x & 31, r = l & 15, h = l >> 4;
  const float* p = base + (size_t)r * ld + 8 * h;
  const float* s = sk + 8 * h;
  bf16x16 f;
#pragma unroll
  for (int e = 0; e < 8; ++e) { f[e] = (__bf16)(p[e] * s[e]); f[e + 8] = (__bf16)(p[e + 16] * s[e + 16]); }
  return f;
}
// A where A[r][k] = base[k*ld + r]   (column-major source)
__device__ __forceinline__ bf16x16 frag_a_col(const float* __restrict__ base, int ld) {
  const int l = threadIdx.x & 31, r = l & 15, h = l >> 4;
  const float* p = base + (size_t)(8 * h) * ld + r;
  bf16x16 f;
#pragma unroll
  for (int e = 0; e < 8; ++e) { f[e] = (__bf16)p[(size_t)e * ld]; f[e + 8] = (__bf16)p[(size_t)(e + 16) * ld]; }
  return f;
}
// B where B[k][c] = base[k*ld + c]   (row-major B, per-element 16-lane coalesced)
__device__ __forceinline__ bf16x16 frag_b_row(const float* __restrict__ base, int ld) {
  const int l = threadIdx.x & 31, c = l & 15, h = l >> 4;
  const float* p = base + (size_t)(8 * h) * ld + c;
  bf16x16 f;
#pragma unroll
  for (int e = 0; e < 16; ++e) f[e] = (__bf16)p[(size_t)(e < 8 ? e : e + 8) * ld];
  return f;
}
// B where B[k][c] = base[c*ld + k]   (per-lane contiguous)
__device__ __forceinline__ bf16x16 frag_b_col(const float* __restrict__ base, int ld) {
  const int l = threadIdx.x & 31, c = l & 15, h = l >> 4;
  const float* p = base + (size_t)c * ld + 8 * h;
  bf16x16 f;
#pragma unroll
  for (int e = 0; e < 16; ++e) f[e] = (__bf16)p[e < 8 ? e : e + 8];
  return f;
}
// same but bf16 LDS source
__device__ __forceinline__ bf16x16 frag_b_col_lds(const __bf16* base, int ld) {
  const int l = threadIdx.x & 31, c = l & 15, h = l >> 4;
  const __bf16* p = base + c * ld + 8 * h;
  bf16x16 f;
#pragma unroll
  for (int e = 0; e < 16; ++e) f[e] = p[e < 8 ? e : e + 8];
  return f;
}

__device__ __forceinline__ float gelu_f(float x) {
  const float k0 = 0.7978845608028654f, k1 = 0.044715f;
  return 0.5f * x * (1.0f + tanhf(k0 * (x + k1 * x * x * x)));
}

// ---------------- K0: zero ----------------
__global__ void k_zero(float* __restrict__ p, int n) {
  int i = blockIdx.x * 256 + threadIdx.x;
  if (i < n) p[i] = 0.0f;
}

// ---------------- K1: forward transform  xhat[c][s] += x[c,hw]*basis[hw,s] ----------------
// grid (S/64=9, B, 4 k-parts), block 256 (8 waves). LDS-transposed basis chunks, atomic reduce.
#define LDB1 40
__global__ void k_fwd(const float* __restrict__ x, const float* __restrict__ br,
                      const float* __restrict__ bi, float* __restrict__ xr, float* __restrict__ xi) {
  const int sb = blockIdx.x * 64, b = blockIdx.y, kp = blockIdx.z;
  const int wave = threadIdx.x >> 5;
  __shared__ __bf16 lr[64 * LDB1], li[64 * LDB1];
  const float* xb  = x  + (size_t)b * C_ * HW_;
  const float* brb = br + (size_t)b * HW_ * S_;
  const float* bib = bi + (size_t)b * HW_ * S_;
  f32x8 accr[4] = {}, acci[4] = {};
  const int c0 = wave * 16;
  const int k0 = kp * (HW_ / 4), k1 = k0 + (HW_ / 4);
  for (int kc = k0; kc < k1; kc += 32) {
    __syncthreads();
    for (int idx = threadIdx.x; idx < 32 * 64; idx += 256) {
      int r = idx >> 6, cc = idx & 63;
      lr[cc * LDB1 + r] = (__bf16)brb[(size_t)(kc + r) * S_ + sb + cc];
      li[cc * LDB1 + r] = (__bf16)bib[(size_t)(kc + r) * S_ + sb + cc];
    }
    __syncthreads();
    bf16x16 a = frag_a_row(xb + (size_t)c0 * HW_ + kc, HW_);
#pragma unroll
    for (int j = 0; j < 4; ++j) {
      bf16x16 fr = frag_b_col_lds(lr + j * 16 * LDB1, LDB1);
      bf16x16 fi = frag_b_col_lds(li + j * 16 * LDB1, LDB1);
      accr[j] = wmma_bf16(a, fr, accr[j]);
      acci[j] = wmma_bf16(a, fi, acci[j]);
    }
  }
  const int l = threadIdx.x & 31, c16 = l & 15, h = l >> 4;
  float* xrb = xr + (size_t)b * C_ * S_;
  float* xib = xi + (size_t)b * C_ * S_;
#pragma unroll
  for (int j = 0; j < 4; ++j)
#pragma unroll
    for (int v = 0; v < 8; ++v) {
      int row = c0 + v + 8 * h, col = sb + 16 * j + c16;
      atomicAdd(&xrb[(size_t)row * S_ + col], accr[j][v]);
      atomicAdd(&xib[(size_t)row * S_ + col], acci[j][v]);
    }
}

// ---------------- K2/K3d: complex projection  out[co,s] = W (cplx) * in[c,s] + bias ----------------
// grid (9, B), block 256.  out_r = Wr*in_r - Wi*in_i + br ; out_i = Wi*in_r + Wr*in_i + bi
__global__ void k_cproj(const float* __restrict__ inr, const float* __restrict__ ini,
                        const float* __restrict__ wr,  const float* __restrict__ wi,
                        const float* __restrict__ bvr, const float* __restrict__ bvi,
                        float* __restrict__ outr, float* __restrict__ outi) {
  const int sb = blockIdx.x * 64, b = blockIdx.y;
  const int wave = threadIdx.x >> 5, co0 = wave * 16;
  const size_t bo = (size_t)b * C_ * S_;
  f32x8 accr[4] = {}, acci[4] = {};
  for (int c = 0; c < C_; c += 32) {
    bf16x16 awr = frag_a_row(wr + (size_t)co0 * C_ + c, C_);
    bf16x16 awi = frag_a_row(wi + (size_t)co0 * C_ + c, C_);
    bf16x16 awn = frag_a_row_neg(wi + (size_t)co0 * C_ + c, C_);
#pragma unroll
    for (int j = 0; j < 4; ++j) {
      bf16x16 tr = frag_b_row(inr + bo + (size_t)c * S_ + sb + 16 * j, S_);
      bf16x16 ti = frag_b_row(ini + bo + (size_t)c * S_ + sb + 16 * j, S_);
      accr[j] = wmma_bf16(awr, tr, accr[j]);
      accr[j] = wmma_bf16(awn, ti, accr[j]);
      acci[j] = wmma_bf16(awi, tr, acci[j]);
      acci[j] = wmma_bf16(awr, ti, acci[j]);
    }
  }
  const int l = threadIdx.x & 31, c16 = l & 15, h = l >> 4;
#pragma unroll
  for (int v = 0; v < 8; ++v) {
    int row = co0 + v + 8 * h;
    float bre = bvr[row], bie = bvi[row];
#pragma unroll
    for (int j = 0; j < 4; ++j) {
      outr[bo + (size_t)row * S_ + sb + 16 * j + c16] = accr[j][v] + bre;
      outi[bo + (size_t)row * S_ + sb + 16 * j + c16] = acci[j][v] + bie;
    }
  }
}

// ---------------- K3a: logits[sq,st] = (qr·kr + qi·ki)/sqrt(dh) ----------------
// grid (36, NH, B), block 128 (4 waves, 9 st-tiles each). K = DH = 32 -> 1 WMMA per term.
__global__ void k_logits(const float* __restrict__ qr, const float* __restrict__ qi,
                         const float* __restrict__ kr, const float* __restrict__ ki,
                         float* __restrict__ logits) {
  const int sq0 = blockIdx.x * 16, hh = blockIdx.y, b = blockIdx.z;
  const int wave = threadIdx.x >> 5;
  const size_t bo = (size_t)b * C_ * S_ + (size_t)hh * DH_ * S_;
  bf16x16 aqr = frag_a_col(qr + bo + sq0, S_);
  bf16x16 aqi = frag_a_col(qi + bo + sq0, S_);
  const float scale = 0.17677669529663687f;  // 1/sqrt(32)
  float* lrow = logits + ((size_t)(b * NH_ + hh) * S_ + sq0) * S_;
  const int l = threadIdx.x & 31, c16 = l & 15, h = l >> 4;
  for (int jt = 0; jt < 9; ++jt) {
    int st0 = (wave * 9 + jt) * 16;
    bf16x16 bkr = frag_b_row(kr + bo + st0, S_);
    bf16x16 bki = frag_b_row(ki + bo + st0, S_);
    f32x8 acc = {};
    acc = wmma_bf16(aqr, bkr, acc);
    acc = wmma_bf16(aqi, bki, acc);
#pragma unroll
    for (int v = 0; v < 8; ++v)
      lrow[(size_t)(v + 8 * h) * S_ + st0 + c16] = acc[v] * scale;
  }
}

// ---------------- K3b: row softmax over 576, one wave32 per row ----------------
__global__ void k_softmax(float* __restrict__ logits) {
  const int wave = threadIdx.x >> 5, l = threadIdx.x & 31;
  const size_t row = (size_t)blockIdx.x * 8 + wave;
  float* p = logits + row * S_;
  float v[18], m = -1e30f;
#pragma unroll
  for (int i = 0; i < 18; ++i) { v[i] = p[l + 32 * i]; m = fmaxf(m, v[i]); }
#pragma unroll
  for (int o = 16; o; o >>= 1) m = fmaxf(m, __shfl_xor(m, o, 32));
  float s = 0.0f;
#pragma unroll
  for (int i = 0; i < 18; ++i) { v[i] = __expf(v[i] - m); s += v[i]; }
#pragma unroll
  for (int o = 16; o; o >>= 1) s += __shfl_xor(s, o, 32);
  float inv = 1.0f / s;
#pragma unroll
  for (int i = 0; i < 18; ++i) p[l + 32 * i] = v[i] * inv;
}

// ---------------- K3c: o = attn · v (complex v) -> (C,S) layout ----------------
// grid (9, NH, B), block 128 (wave = one sq-tile, 2 d-tiles x {r,i})
__global__ void k_attnv(const float* __restrict__ attn, const float* __restrict__ vr,
                        const float* __restrict__ vi, float* __restrict__ outr,
                        float* __restrict__ outi) {
  const int wave = threadIdx.x >> 5;
  const int sq0 = (blockIdx.x * 4 + wave) * 16;
  const int hh = blockIdx.y, b = blockIdx.z;
  const size_t bo = (size_t)b * C_ * S_ + (size_t)hh * DH_ * S_;
  const float* arow = attn + ((size_t)(b * NH_ + hh) * S_ + sq0) * S_;
  f32x8 r0 = {}, r1 = {}, i0 = {}, i1 = {};
  for (int tc = 0; tc < S_; tc += 32) {
    bf16x16 a  = frag_a_row(arow + tc, S_);
    bf16x16 b0 = frag_b_col(vr + bo + tc, S_);
    bf16x16 b1 = frag_b_col(vr + bo + (size_t)16 * S_ + tc, S_);
    bf16x16 c0 = frag_b_col(vi + bo + tc, S_);
    bf16x16 c1 = frag_b_col(vi + bo + (size_t)16 * S_ + tc, S_);
    r0 = wmma_bf16(a, b0, r0); r1 = wmma_bf16(a, b1, r1);
    i0 = wmma_bf16(a, c0, i0); i1 = wmma_bf16(a, c1, i1);
  }
  const int l = threadIdx.x & 31, c16 = l & 15, h = l >> 4;
#pragma unroll
  for (int v = 0; v < 8; ++v) {
    int sq = sq0 + v + 8 * h;
    outr[bo + (size_t)c16 * S_ + sq]              = r0[v];
    outr[bo + (size_t)(16 + c16) * S_ + sq]       = r1[v];
    outi[bo + (size_t)c16 * S_ + sq]              = i0[v];
    outi[bo + (size_t)(16 + c16) * S_ + sq]       = i1[v];
  }
}

// ---------------- K5: inverse transform  x_rec[hw][c] = br·o_r + bi·o_i ----------------
// grid (HW/128=72, B), block 256. o-proj chunks staged to LDS (bf16), basis A-frags direct.
#define LDB2 34
__global__ void k_inv(const float* __restrict__ br, const float* __restrict__ bi,
                      const float* __restrict__ opr, const float* __restrict__ opi,
                      float* __restrict__ xrec) {
  const int b = blockIdx.y, wave = threadIdx.x >> 5;
  const int hw0 = blockIdx.x * 128 + wave * 16;
  const size_t bbo = (size_t)b * HW_ * S_;
  const size_t obo = (size_t)b * C_ * S_;
  __shared__ __bf16 lpr[C_ * LDB2], lpi[C_ * LDB2];
  f32x8 acc[8] = {};
  for (int sc = 0; sc < S_; sc += 32) {
    __syncthreads();
    for (int idx = threadIdx.x; idx < C_ * 32; idx += 256) {
      int cc = idx >> 5, k = idx & 31;
      lpr[cc * LDB2 + k] = (__bf16)opr[obo + (size_t)cc * S_ + sc + k];
      lpi[cc * LDB2 + k] = (__bf16)opi[obo + (size_t)cc * S_ + sc + k];
    }
    __syncthreads();
    bf16x16 ar = frag_a_row(br + bbo + (size_t)hw0 * S_ + sc, S_);
    bf16x16 ai = frag_a_row(bi + bbo + (size_t)hw0 * S_ + sc, S_);
#pragma unroll
    for (int t = 0; t < 8; ++t) {
      bf16x16 fr = frag_b_col_lds(lpr + t * 16 * LDB2, LDB2);
      bf16x16 fi = frag_b_col_lds(lpi + t * 16 * LDB2, LDB2);
      acc[t] = wmma_bf16(ar, fr, acc[t]);
      acc[t] = wmma_bf16(ai, fi, acc[t]);
    }
  }
  const int l = threadIdx.x & 31, c16 = l & 15, h = l >> 4;
  float* xo = xrec + (size_t)b * HW_ * C_;
#pragma unroll
  for (int t = 0; t < 8; ++t)
#pragma unroll
    for (int v = 0; v < 8; ++v)
      xo[(size_t)(hw0 + v + 8 * h) * C_ + 16 * t + c16] = acc[t][v];
}

// ---------------- K4a: (x_rec*alpha) @ mixer_w^T + b -> LayerNorm -> gelu -> tmp (B,HW,C) ----------------
// grid (HW/64=144, B), block 128 (4 waves, 16 hw-rows each, full 128 co)
#define LDY 130
__global__ void k_mixer(const float* __restrict__ xrec, const float* __restrict__ mw,
                        const float* __restrict__ mb, const float* __restrict__ alpha,
                        const float* __restrict__ gamma, const float* __restrict__ beta,
                        float* __restrict__ tmp) {
  const int b = blockIdx.y, hwb = blockIdx.x * 64;
  const int wave = threadIdx.x >> 5;
  __shared__ float y[64 * LDY];
  __shared__ float al[C_], mbs[C_], gs[C_], bs[C_], mus[64], rss[64];
  al[threadIdx.x]  = alpha[threadIdx.x];
  mbs[threadIdx.x] = mb[threadIdx.x];
  gs[threadIdx.x]  = gamma[threadIdx.x];
  bs[threadIdx.x]  = beta[threadIdx.x];
  __syncthreads();
  const size_t xo = (size_t)b * HW_ * C_ + (size_t)(hwb + wave * 16) * C_;
  f32x8 acc[8] = {};
  for (int c = 0; c < C_; c += 32) {
    bf16x16 a = frag_a_row_scaled(xrec + xo + c, C_, al + c);
#pragma unroll
    for (int t = 0; t < 8; ++t) {
      bf16x16 bm = frag_b_col(mw + (size_t)(t * 16) * C_ + c, C_);
      acc[t] = wmma_bf16(a, bm, acc[t]);
    }
  }
  const int l = threadIdx.x & 31, c16 = l & 15, h = l >> 4;
#pragma unroll
  for (int t = 0; t < 8; ++t)
#pragma unroll
    for (int v = 0; v < 8; ++v)
      y[(wave * 16 + v + 8 * h) * LDY + 16 * t + c16] = acc[t][v] + mbs[16 * t + c16];
  __syncthreads();
  if (threadIdx.x < 64) {
    float s = 0.0f, q = 0.0f;
    for (int cc = 0; cc < C_; ++cc) { float z = y[threadIdx.x * LDY + cc]; s += z; q += z * z; }
    float mu = s * (1.0f / C_);
    float var = q * (1.0f / C_) - mu * mu;
    mus[threadIdx.x] = mu;
    rss[threadIdx.x] = rsqrtf(var + 1e-5f);
  }
  __syncthreads();
  for (int r = 0; r < 64; ++r) {
    float z = y[r * LDY + threadIdx.x];
    z = (z - mus[r]) * rss[r] * gs[threadIdx.x] + bs[threadIdx.x];
    tmp[(size_t)b * HW_ * C_ + (size_t)(hwb + r) * C_ + threadIdx.x] = gelu_f(z);
  }
}

// ---------------- K4b: shortcut GEMM + add tmp (transposed via LDS) + gelu -> out (B,C,HW) ----------------
// grid (HW/64=144, B), block 256 (8 waves: 16 co-rows each x 64 hw)
__global__ void k_shortcut(const float* __restrict__ x, const float* __restrict__ sw,
                           const float* __restrict__ sb, const float* __restrict__ tmp,
                           float* __restrict__ out) {
  const int b = blockIdx.y, hwb = blockIdx.x * 64;
  const int wave = threadIdx.x >> 5, co0 = wave * 16;
  __shared__ float tl[64 * LDY];
  for (int idx = threadIdx.x; idx < 64 * C_; idx += 256) {
    int r = idx >> 7, cc = idx & 127;
    tl[r * LDY + cc] = tmp[(size_t)b * HW_ * C_ + (size_t)(hwb + r) * C_ + cc];
  }
  __syncthreads();
  f32x8 acc[4] = {};
  for (int c = 0; c < C_; c += 32) {
    bf16x16 a = frag_a_row(sw + (size_t)co0 * C_ + c, C_);
#pragma unroll
    for (int t = 0; t < 4; ++t) {
      bf16x16 bx = frag_b_row(x + (size_t)b * C_ * HW_ + (size_t)c * HW_ + hwb + 16 * t, HW_);
      acc[t] = wmma_bf16(a, bx, acc[t]);
    }
  }
  const int l = threadIdx.x & 31, c16 = l & 15, h = l >> 4;
#pragma unroll
  for (int t = 0; t < 4; ++t)
#pragma unroll
    for (int v = 0; v < 8; ++v) {
      int co = co0 + v + 8 * h, hwl = 16 * t + c16;
      float val = acc[t][v] + sb[co] + tl[hwl * LDY + co];
      out[(size_t)b * C_ * HW_ + (size_t)co * HW_ + hwb + hwl] = gelu_f(val);
    }
}

// ---------------- launch ----------------
extern "C" void kernel_launch(void* const* d_in, const int* in_sizes, int n_in,
                              void* d_out, int out_size, void* d_ws, size_t ws_size,
                              hipStream_t stream) {
  (void)in_sizes; (void)n_in; (void)out_size; (void)ws_size;
  const float* x    = (const float*)d_in[0];
  const float* bsr  = (const float*)d_in[1];
  const float* bsi  = (const float*)d_in[2];
  const float* awr  = (const float*)d_in[3];
  const float* awi  = (const float*)d_in[4];
  const float* abr  = (const float*)d_in[5];
  const float* abi  = (const float*)d_in[6];
  const float* alp  = (const float*)d_in[7];
  const float* mw   = (const float*)d_in[8];
  const float* mb   = (const float*)d_in[9];
  const float* gam  = (const float*)d_in[10];
  const float* bet  = (const float*)d_in[11];
  const float* sw   = (const float*)d_in[12];
  const float* sbv  = (const float*)d_in[13];
  float* out = (float*)d_out;
  float* ws  = (float*)d_ws;

  const size_t NA = (size_t)B_ * C_ * S_;          // 589,824 floats per (C,S) array
  const size_t LG = (size_t)B_ * NH_ * S_ * S_;    // 10,616,832 floats
  float* xr  = ws + 0 * NA;
  float* xi  = ws + 1 * NA;
  float* qr  = ws + 2 * NA;
  float* qi  = ws + 3 * NA;
  float* kr  = ws + 4 * NA;
  float* ki  = ws + 5 * NA;
  float* vr  = ws + 6 * NA;
  float* vi  = ws + 7 * NA;
  float* orr = ws + 8 * NA;
  float* oii = ws + 9 * NA;
  float* opr = ws + 10 * NA;
  float* opi = ws + 11 * NA;
  float* logits = ws + 12 * NA;
  float* xrec   = logits;                          // alias: logits dead once attn·v done
  float* tmp    = ws + 12 * NA + LG;

  const int CC = C_ * C_;
  // 0) zero xhat accumulators
  {
    int n = (int)(2 * NA);
    k_zero<<<(n + 255) / 256, 256, 0, stream>>>(xr, n);
  }
  // 1) forward transform (K split 4x, atomic reduce)
  k_fwd<<<dim3(S_ / 64, B_, 4), 256, 0, stream>>>(x, bsr, bsi, xr, xi);
  // 2) q, k, v complex projections
  k_cproj<<<dim3(S_ / 64, B_), 256, 0, stream>>>(xr, xi, awr + 0 * CC, awi + 0 * CC,
                                                 abr + 0 * C_, abi + 0 * C_, qr, qi);
  k_cproj<<<dim3(S_ / 64, B_), 256, 0, stream>>>(xr, xi, awr + 1 * CC, awi + 1 * CC,
                                                 abr + 1 * C_, abi + 1 * C_, kr, ki);
  k_cproj<<<dim3(S_ / 64, B_), 256, 0, stream>>>(xr, xi, awr + 2 * CC, awi + 2 * CC,
                                                 abr + 2 * C_, abi + 2 * C_, vr, vi);
  // 3) attention logits, softmax, attn·v
  k_logits<<<dim3(S_ / 16, NH_, B_), 128, 0, stream>>>(qr, qi, kr, ki, logits);
  k_softmax<<<dim3((B_ * NH_ * S_) / 8), 256, 0, stream>>>(logits);
  k_attnv<<<dim3(S_ / 64, NH_, B_), 128, 0, stream>>>(logits, vr, vi, orr, oii);
  // 4) output projection
  k_cproj<<<dim3(S_ / 64, B_), 256, 0, stream>>>(orr, oii, awr + 3 * CC, awi + 3 * CC,
                                                 abr + 3 * C_, abi + 3 * C_, opr, opi);
  // 5) inverse transform -> x_rec (B,HW,C)
  k_inv<<<dim3(HW_ / 128, B_), 256, 0, stream>>>(bsr, bsi, opr, opi, xrec);
  // 6) mixer + layernorm + gelu -> tmp (B,HW,C)
  k_mixer<<<dim3(HW_ / 64, B_), 128, 0, stream>>>(xrec, mw, mb, alp, gam, bet, tmp);
  // 7) shortcut GEMM + add + gelu -> out (B,C,H,W)
  k_shortcut<<<dim3(HW_ / 64, B_), 256, 0, stream>>>(x, sw, sbv, tmp, out);
}
